// WindowAttention_82282983457134
// MI455X (gfx1250) — matrix operand order repaired and checked by reading, not compiled, not run on hardware
//
#include <hip/hip_runtime.h>

typedef __attribute__((ext_vector_type(16))) _Float16 v16h;
typedef __attribute__((ext_vector_type(8)))  _Float16 v8h;
typedef __attribute__((ext_vector_type(4)))  _Float16 v4h;
typedef __attribute__((ext_vector_type(8)))  float    v8f;
typedef __attribute__((ext_vector_type(4)))  float    v4f;

#define DIMC   256
#define NHEADS 8
#define NTOK   49
#define NPAD   64
#define HD     32
#define NWIN   4096

// ---------------- WMMA helpers (CDNA5 f16 16x16x32, f32 accum) ----------------

__device__ __forceinline__ v8f wmma_f16(v16h a, v16h b, v8f c) {
  // 8 args: (neg_a, A, neg_b, B, c_mod, C, reuse_a, reuse_b)
  return __builtin_amdgcn_wmma_f32_16x16x32_f16(false, a, false, b, (short)0, c, false, false);
}

__device__ __forceinline__ v16h pack16(v8h lo, v8h hi) {
  union { v16h v; v8h h[2]; } u;
  u.h[0] = lo; u.h[1] = hi;
  return u.v;
}

// A fragment (16x32, MxK) from row-major [m][k] storage, leading dim ld (halfs).
// Lane l (0-15): M=l, K = {0..7, 16..23}; lane 16-31: M=l-16, K = {8..15, 24..31}.
__device__ __forceinline__ v16h load_a(const _Float16* base, int ld, int m0, int k0, int lane) {
  const _Float16* p = base + (m0 + (lane & 15)) * ld + k0 + ((lane >> 4) << 3);
  v8h lo = *(const v8h*)p;        // K = kh .. kh+7
  v8h hi = *(const v8h*)(p + 16); // K = 16+kh .. 16+kh+7
  return pack16(lo, hi);
}

// B fragment (32x16, KxN) from N-major (transposed) storage bt[n][k], leading dim ld.
// Lane l (0-15): N=l, K = {0..7, 16..23}; lane 16-31: N=l-16, K = {8..15, 24..31}.
__device__ __forceinline__ v16h load_b(const _Float16* bt, int ld, int n0, int k0, int lane) {
  const _Float16* p = bt + (n0 + (lane & 15)) * ld + k0 + ((lane >> 4) << 3);
  v8h lo = *(const v8h*)p;
  v8h hi = *(const v8h*)(p + 16);
  return pack16(lo, hi);
}

// ---------------- Prep: transpose weights to f16 N-major, bias in C-fragment order ----------------

__global__ void __launch_bounds__(256) prep_kernel(
    const float* __restrict__ qkv_w,      // [256][768]
    const float* __restrict__ proj_w,     // [256][256]
    const float* __restrict__ bias_table, // [169][8]
    const int*   __restrict__ rel_idx,    // [49][49]
    _Float16* __restrict__ qkv_wt,        // [768][256] f16
    _Float16* __restrict__ proj_wt,       // [256][256] f16
    float*    __restrict__ biasFrag)      // [8][4 mt][4 nt][32 lane][8 r] f32
{
  int idx = blockIdx.x * 256 + threadIdx.x;
  if (idx < 768 * 256) {
    int n = idx >> 8, k = idx & 255;
    qkv_wt[idx] = (_Float16)qkv_w[k * 768 + n];
  } else if (idx < 768 * 256 + 256 * 256) {
    int i = idx - 768 * 256;
    int n = i >> 8, k = i & 255;
    proj_wt[i] = (_Float16)proj_w[k * 256 + n];
  } else if (idx < 768 * 256 + 256 * 256 + NHEADS * 4096) {
    int i = idx - (768 * 256 + 256 * 256);
    // decompose fragment-order index: [h][mt][nt][lane][r]
    int r    = i & 7;
    int lane = (i >> 3) & 31;
    int nt   = (i >> 8) & 3;
    int mt   = (i >> 10) & 3;
    int h    = i >> 12;
    // C-tile layout: row m = mt*16 + (lane>=16 ? 8 : 0) + r, col n = nt*16 + (lane&15)
    int m = (mt << 4) + ((lane >> 4) << 3) + r;
    int n = (nt << 4) + (lane & 15);
    float v;
    if (m < NTOK && n < NTOK)      v = bias_table[rel_idx[m * NTOK + n] * NHEADS + h];
    else if (m < NTOK)             v = -1e30f;   // mask padded key columns -> exp = 0
    else                           v = 0.0f;     // padded query rows: finite uniform softmax
    biasFrag[i] = v;
  }
}

// ---------------- Fused window attention: one workgroup per window, 8 waves ----------------
// LDS layout (halfs):
//   [0,16384)       Xh  [64][256]         (phase 1)   -- aliased by:
//   [0,16384)       Oh  [64][256]         (phase 2/3)
//   [16384,25600)   P   [8 waves][16][72] (phase 2)
//   [25600,41984)   Q   [8][64][32]
//   [41984,58368)   K   [8][64][32]
//   [58368,76800)   Vt  [8][32][72]
// total = 76800 halfs = 153600 B -> 2 workgroups per WGP (320 KB LDS)

__global__ void __launch_bounds__(256) win_attn_kernel(
    const float*    __restrict__ x,        // [4096][49][256]
    const _Float16* __restrict__ qkv_wt,   // [768][256]
    const _Float16* __restrict__ proj_wt,  // [256][256]
    const float*    __restrict__ biasFrag, // [8][4][4][32][8]
    const float*    __restrict__ proj_b,   // [256]
    float*          __restrict__ out)      // [4096][49][256]
{
  extern __shared__ _Float16 smem[];
  _Float16* Xh  = smem;
  _Float16* Oh  = smem;
  _Float16* Pb  = smem + 16384;
  _Float16* Qs  = smem + 25600;
  _Float16* Ks  = smem + 41984;
  _Float16* Vts = smem + 58368;

  const int tid  = threadIdx.x;
  const int lane = tid & 31;
  const int wave = tid >> 5;
  const int col  = lane & 15;        // N-position within a 16x16 C tile
  const int rhi  = (lane >> 4) << 3; // row offset 0 / 8 within a C tile
  const int b    = blockIdx.x;

  // ---- Phase 1: x -> f16 LDS, pad rows 49..63 with zero ----
  {
    const float4* xb = (const float4*)(x + (size_t)b * NTOK * DIMC);
    for (int i = tid; i < NTOK * DIMC / 4; i += 256) {
      float4 v = xb[i];
      v4h hv;
      hv[0] = (_Float16)v.x; hv[1] = (_Float16)v.y;
      hv[2] = (_Float16)v.z; hv[3] = (_Float16)v.w;
      *(v4h*)(Xh + i * 4) = hv;
    }
    v4h z = {};
    for (int i = NTOK * DIMC / 4 + tid; i < NPAD * DIMC / 4; i += 256)
      *(v4h*)(Xh + i * 4) = z;
  }
  __syncthreads();

  // ---- Phase 2: qkv = Xh(64x256) @ qkv_wt -> Q (pre-scaled), K, V-transposed in LDS ----
  const float qscale = 0.17677669529663687f; // 1/sqrt(32)
  for (int i = 0; i < 6; ++i) {
    int nt = wave + (i << 3);   // 48 N-tiles over 8 waves
    int nc = nt << 4;
    v8f acc[4] = {};
#pragma unroll
    for (int kt = 0; kt < 8; ++kt) {
      v16h bf = load_b(qkv_wt + (size_t)nc * DIMC, DIMC, 0, kt << 5, lane);
#pragma unroll
      for (int mt = 0; mt < 4; ++mt) {
        v16h af = load_a(Xh, DIMC, mt << 4, kt << 5, lane);
        acc[mt] = wmma_f16(af, bf, acc[mt]);
      }
    }
    int s = nc >> 8, rem = nc & 255, h = rem >> 5, dp = rem & 31;
    _Float16* qdst = Qs  + h * (NPAD * HD);
    _Float16* kdst = Ks  + h * (NPAD * HD);
    _Float16* vdst = Vts + h * (HD * 72);
#pragma unroll
    for (int mt = 0; mt < 4; ++mt) {
#pragma unroll
      for (int r = 0; r < 8; ++r) {
        int m = (mt << 4) + rhi + r;
        float v = acc[mt][r];
        if (s == 0)      qdst[m * HD + dp + col] = (_Float16)(v * qscale);
        else if (s == 1) kdst[m * HD + dp + col] = (_Float16)v;
        else             vdst[(dp + col) * 72 + m] = (_Float16)v; // V transposed [d][tok]
      }
    }
  }
  __syncthreads();

  // ---- Phase 3: per-head attention (head == wave), softmax in registers ----
  {
    const int h = wave;
    const _Float16* Qh = Qs  + h * (NPAD * HD);
    const _Float16* Kh = Ks  + h * (NPAD * HD);
    const _Float16* Vh = Vts + h * (HD * 72);
    _Float16* Pw = Pb + wave * (16 * 72);
    const float* bb = biasFrag + h * 4096 + (lane << 3);

    for (int mt = 0; mt < 4; ++mt) {
      // bias preloaded as the WMMA accumulator (C matrix), fragment-ordered:
      // two contiguous 16B loads per tile, fully coalesced across the wave
      v8f sc[4];
#pragma unroll
      for (int nt = 0; nt < 4; ++nt) {
        const float* bp = bb + (((mt << 2) + nt) << 8);
        v4f lo = *(const v4f*)bp;
        v4f hi = *(const v4f*)(bp + 4);
#pragma unroll
        for (int r = 0; r < 4; ++r) { sc[nt][r] = lo[r]; sc[nt][4 + r] = hi[r]; }
      }
      // S = (scale*Q) @ K^T + bias : single K=32 WMMA per tile
      v16h aq = load_a(Qh, HD, mt << 4, 0, lane);
#pragma unroll
      for (int nt = 0; nt < 4; ++nt) {
        v16h bk = load_b(Kh, HD, nt << 4, 0, lane);
        sc[nt] = wmma_f16(aq, bk, sc[nt]);
      }
      // softmax per query row; rows live across the 16-lane group (xor 1/2/4/8 stays inside it)
#pragma unroll
      for (int r = 0; r < 8; ++r) {
        float mx = fmaxf(fmaxf(sc[0][r], sc[1][r]), fmaxf(sc[2][r], sc[3][r]));
        mx = fmaxf(mx, __shfl_xor(mx, 1));
        mx = fmaxf(mx, __shfl_xor(mx, 2));
        mx = fmaxf(mx, __shfl_xor(mx, 4));
        mx = fmaxf(mx, __shfl_xor(mx, 8));
        float sum = 0.f;
#pragma unroll
        for (int nt = 0; nt < 4; ++nt) {
          float e = __expf(sc[nt][r] - mx);
          sc[nt][r] = e; sum += e;
        }
        sum += __shfl_xor(sum, 1);
        sum += __shfl_xor(sum, 2);
        sum += __shfl_xor(sum, 4);
        sum += __shfl_xor(sum, 8);
        float inv = 1.0f / sum;
#pragma unroll
        for (int nt = 0; nt < 4; ++nt)
          Pw[(rhi + r) * 72 + (nt << 4) + col] = (_Float16)(sc[nt][r] * inv);
      }
      // O[mt] = P(16x64) @ V(64x32)
      v8f o0 = {}, o1 = {};
#pragma unroll
      for (int kt = 0; kt < 2; ++kt) {
        v16h ap = load_a(Pw, 72, 0, kt << 5, lane);
        o0 = wmma_f16(ap, load_b(Vh, 72, 0,  kt << 5, lane), o0);
        o1 = wmma_f16(ap, load_b(Vh, 72, 16, kt << 5, lane), o1);
      }
#pragma unroll
      for (int r = 0; r < 8; ++r) {
        int m = (mt << 4) + rhi + r;
        Oh[m * DIMC + h * HD + col]      = (_Float16)o0[r];
        Oh[m * DIMC + h * HD + 16 + col] = (_Float16)o1[r];
      }
    }
  }
  __syncthreads();

  // ---- Phase 4: out = Oh(64x256) @ proj_wt + proj_b, store valid rows ----
  {
    float* outb = out + (size_t)b * NTOK * DIMC;
#pragma unroll
    for (int j = 0; j < 2; ++j) {
      int nc = (((wave << 1) + j) << 4);
      v8f acc[4] = {};
#pragma unroll
      for (int kt = 0; kt < 8; ++kt) {
        v16h bf = load_b(proj_wt + (size_t)nc * DIMC, DIMC, 0, kt << 5, lane);
#pragma unroll
        for (int mt = 0; mt < 4; ++mt) {
          v16h af = load_a(Oh, DIMC, mt << 4, kt << 5, lane);
          acc[mt] = wmma_f16(af, bf, acc[mt]);
        }
      }
      float pb = proj_b[nc + col];
#pragma unroll
      for (int mt = 0; mt < 4; ++mt) {
#pragma unroll
        for (int r = 0; r < 8; ++r) {
          int m = (mt << 4) + rhi + r;
          if (m < NTOK) outb[m * DIMC + nc + col] = acc[mt][r] + pb;
        }
      }
    }
  }
}

// ---------------- Launch ----------------

extern "C" void kernel_launch(void* const* d_in, const int* in_sizes, int n_in,
                              void* d_out, int out_size, void* d_ws, size_t ws_size,
                              hipStream_t stream) {
  const float* x          = (const float*)d_in[0];
  const float* qkv_w      = (const float*)d_in[1];
  const float* proj_w     = (const float*)d_in[2];
  const float* proj_b     = (const float*)d_in[3];
  const float* bias_table = (const float*)d_in[4];
  const int*   rel_idx    = (const int*)d_in[5];

  // workspace: qkv_wt f16 (384 KB) | proj_wt f16 (128 KB) | biasFrag f32 (128 KB)
  _Float16* qkv_wt   = (_Float16*)d_ws;
  _Float16* proj_wt  = (_Float16*)((char*)d_ws + 768 * 256 * 2);
  float*    biasFrag = (float*)((char*)d_ws + 768 * 256 * 2 + 256 * 256 * 2);

  int prep_total = 768 * 256 + 256 * 256 + NHEADS * 4096;
  prep_kernel<<<(prep_total + 255) / 256, 256, 0, stream>>>(
      qkv_w, proj_w, bias_table, rel_idx, qkv_wt, proj_wt, biasFrag);

  size_t smem_bytes = 76800 * sizeof(_Float16); // 153600 B
  win_attn_kernel<<<NWIN, 256, smem_bytes, stream>>>(
      x, qkv_wt, proj_wt, biasFrag, proj_b, (float*)d_out);
}
